// IntegratedDyRepLayer_15401752723644
// MI455X (gfx1250) — matrix operand order repaired and checked by reading, not compiled, not run on hardware
//
#include <hip/hip_runtime.h>
#include <math.h>

typedef _Float16 h16;
typedef __attribute__((ext_vector_type(16))) _Float16 v16h;
typedef __attribute__((ext_vector_type(8)))  _Float16 v8h;
typedef __attribute__((ext_vector_type(4)))  _Float16 v4h;
typedef __attribute__((ext_vector_type(8)))  float    v8f;
typedef __attribute__((ext_vector_type(4)))  float    v4f;

#define BATCH   131072
#define MEMD    128
#define EDGED   172
#define TIMED   100
#define PITCH   576            // act row pitch in halves (mult of 32)
#define ROWS_W  16             // batch rows per wave
#define WPB     4              // waves per block
#define ACT_H   (ROWS_W*PITCH) // halves per activation buffer (9216)

// fixed K-offsets inside the activation buffer
#define OFF_EDGE 0
#define OFF_TE   172
#define OFF_A    272           // slot A: mem / evo_self / assoc / h / upd
#define OFF_B    400           // slot B: evo_other / embeddings

// extended-weight K windows (32-aligned)
#define EVO_KMIN 160
#define EVO_KWIN 256
#define ASSOC_KMIN 0
#define ASSOC_KWIN 544
#define C1_KMIN 0
#define C1_KWIN 416
#define C2_KMIN 272
#define C2_KWIN 128
#define OUT_KMIN 272
#define OUT_KWIN 256

// workspace layout (offsets in halves)
#define WS_EVO   0
#define WS_ASSOC 32768
#define WS_C1    102400
#define WS_C2    155648
#define WS_OUT   172032   // total 204800 halves = 409600 bytes

struct Seg { int k0, k1, src; };

// ---- fast transcendentals: keep epilogues branch-free and in the TRANS pipe ----
__device__ __forceinline__ float fast_tanh(float x) {
#if __has_builtin(__builtin_amdgcn_tanhf)
  return __builtin_amdgcn_tanhf(x);          // v_tanh_f32
#else
  float xc = fminf(fmaxf(x, -15.0f), 15.0f);
  float e  = __builtin_amdgcn_exp2f(xc * 2.885390081777927f);  // e^(2x)
  return (e - 1.0f) * __builtin_amdgcn_rcpf(e + 1.0f);
#endif
}

__device__ __forceinline__ float fast_cos(float x) {
  // v_cos_f32 takes revolutions (x / 2pi), same contract as OCML native_cos
  return __builtin_amdgcn_cosf(x * 0.15915494309189535f);
}

// ---- prolog: convert fp32 weight [fan_in][128] -> f16 col-major extended window ----
__global__ void pack_w(const float* __restrict__ W, h16* __restrict__ dst,
                       int kwin, int kmin, Seg s0, Seg s1, Seg s2, Seg s3) {
  int idx = blockIdx.x * 256 + threadIdx.x;
  int total = 128 * kwin;
  if (idx >= total) return;
  int col  = idx / kwin;
  int krel = idx - col * kwin;
  int k    = kmin + krel;
  float v = 0.0f;
  if      (k >= s0.k0 && k < s0.k1) v = W[(s0.src + k - s0.k0) * 128 + col];
  else if (k >= s1.k0 && k < s1.k1) v = W[(s1.src + k - s1.k0) * 128 + col];
  else if (k >= s2.k0 && k < s2.k1) v = W[(s2.src + k - s2.k0) * 128 + col];
  else if (k >= s3.k0 && k < s3.k1) v = W[(s3.src + k - s3.k0) * 128 + col];
  dst[col * kwin + krel] = (h16)v;
}

// ---- fragment helpers (contiguous k = 16*half + i in both A and B, 16-bit A 16x32 / B 32x16) ----
__device__ __forceinline__ v8f vzero8() { v8f z = {}; return z; }

__device__ __forceinline__ v4h cvt4(v4f x) {
  v4h r; r.x = (h16)x.x; r.y = (h16)x.y; r.z = (h16)x.z; r.w = (h16)x.w; return r;
}

__device__ __forceinline__ v16h a_frag(const h16* act, int mrow, int half, int kb) {
  // lane(row=mrow, half): 16 contiguous K values K[kb+half*16 .. +15]
  const h16* p = act + mrow * PITCH + kb + half * 16;
  v8h lo = *(const v8h*)p;
  v8h hi = *(const v8h*)(p + 8);
  v16h a;
#pragma unroll
  for (int i = 0; i < 8; ++i) { a[i] = lo[i]; a[i + 8] = hi[i]; }
  return a;
}

__device__ __forceinline__ v16h b_frag(const h16* __restrict__ w, int colBase, int kRel) {
  // lane(col, half): 16 contiguous K values starting at kRel (= kb-kmin + half*16)
  const h16* p = w + colBase + kRel;
  v8h lo = *(const v8h*)p;
  v8h hi = *(const v8h*)(p + 8);
  v16h b;
#pragma unroll
  for (int i = 0; i < 8; ++i) { b[i] = lo[i]; b[i + 8] = hi[i]; }
  return b;
}

// dual-branch GEMM: one B-fragment load feeds src and dst WMMAs
__device__ __forceinline__ void gemm2(const h16* actS, const h16* actD,
                                      const h16* __restrict__ w,
                                      int kmin, int kmax, int kwin,
                                      int mrow, int half,
                                      v8f (&cS)[8], v8f (&cD)[8]) {
#pragma unroll
  for (int nb = 0; nb < 8; ++nb) { cS[nb] = vzero8(); cD[nb] = vzero8(); }
#pragma unroll 1
  for (int kb = kmin; kb < kmax; kb += 32) {
    v16h aS = a_frag(actS, mrow, half, kb);
    v16h aD = a_frag(actD, mrow, half, kb);
    int kRel = (kb - kmin) + half * 16;
#pragma unroll
    for (int nb = 0; nb < 8; ++nb) {
      v16h b = b_frag(w, (nb * 16 + mrow) * kwin, kRel);
      cS[nb] = __builtin_amdgcn_wmma_f32_16x16x32_f16(false, aS, false, b,
                                                      (short)0, cS[nb], false, false);
      cD[nb] = __builtin_amdgcn_wmma_f32_16x16x32_f16(false, aD, false, b,
                                                      (short)0, cD[nb], false, false);
    }
  }
}

// ---- main fused kernel: 1 wave = 16 batch rows, both branches ----
__global__ __launch_bounds__(WPB * 32) void dyrep_main(
    const float* __restrict__ src_emb, const float* __restrict__ dst_emb,
    const int*   __restrict__ src_ids, const int*   __restrict__ dst_ids,
    const float* __restrict__ edge,    const float* __restrict__ ts,
    const float* __restrict__ memory,
    const float* __restrict__ time_w,  const float* __restrict__ time_b,
    const float* __restrict__ evo_b,   const float* __restrict__ assoc_b,
    const float* __restrict__ comm_b1, const float* __restrict__ comm_b2,
    const float* __restrict__ out_b,
    const h16* __restrict__ w_evo, const h16* __restrict__ w_assoc,
    const h16* __restrict__ w_c1,  const h16* __restrict__ w_c2,
    const h16* __restrict__ w_out,
    float* __restrict__ out) {
  extern __shared__ __align__(16) h16 smem[];
  const int wave = threadIdx.x >> 5;
  const int lane = threadIdx.x & 31;
  h16* actS = smem + wave * (2 * ACT_H);
  h16* actD = actS + ACT_H;
  const int row0 = (blockIdx.x * WPB + wave) * ROWS_W;
  const int mrow = lane & 15;   // A row / B col-in-block for this lane
  const int half = lane >> 4;

  v8f cS[8], cD[8];

  // zero both act buffers (covers padding / not-yet-written slots)
  for (int i = lane; i < (2 * ACT_H) / 8; i += 32) {
    v8h z = {};
    *(v8h*)(actS + i * 8) = z;
  }

  // time encoding: te = cos(t * w + b), written to both branch buffers
  for (int idx = lane; idx < ROWS_W * TIMED; idx += 32) {
    int r = idx / TIMED;
    int j = idx - r * TIMED;
    float t = ts[row0 + r];
    h16 hv = (h16)fast_cos(t * time_w[j] + time_b[j]);
    actS[r * PITCH + OFF_TE + j] = hv;
    actD[r * PITCH + OFF_TE + j] = hv;
  }

  // edge features
  for (int idx = lane; idx < ROWS_W * EDGED; idx += 32) {
    int r = idx / EDGED;
    int c = idx - r * EDGED;
    h16 hv = (h16)edge[(size_t)(row0 + r) * EDGED + c];
    actS[r * PITCH + OFF_EDGE + c] = hv;
    actD[r * PITCH + OFF_EDGE + c] = hv;
  }

  // memory gather -> slot A
  for (int q = lane; q < ROWS_W * (MEMD / 4); q += 32) {
    int r  = q >> 5;              // 32 quads per row
    int c4 = (q & 31) * 4;
    long sid = src_ids[row0 + r];
    long did = dst_ids[row0 + r];
    v4f ms = *(const v4f*)(memory + sid * MEMD + c4);
    v4f md = *(const v4f*)(memory + did * MEMD + c4);
    *(v4h*)(actS + r * PITCH + OFF_A + c4) = cvt4(ms);
    *(v4h*)(actD + r * PITCH + OFF_A + c4) = cvt4(md);
  }

  // ---- stage 1: evolution  evo = tanh([mem|te] @ W) ----
  gemm2(actS, actD, w_evo, EVO_KMIN, EVO_KMIN + EVO_KWIN, EVO_KWIN, mrow, half, cS, cD);
#pragma unroll
  for (int nb = 0; nb < 8; ++nb) {
    int col = nb * 16 + mrow;
    float bv = evo_b[col];
#pragma unroll
    for (int v = 0; v < 8; ++v) {
      int ro = v + half * 8;
      h16 eS = (h16)fast_tanh(cS[nb][v] + bv);
      h16 eD = (h16)fast_tanh(cD[nb][v] + bv);
      actS[ro * PITCH + OFF_A + col] = eS;   // self for src branch
      actD[ro * PITCH + OFF_B + col] = eS;   // other for dst branch (also persists evo_src)
      actD[ro * PITCH + OFF_A + col] = eD;
      actS[ro * PITCH + OFF_B + col] = eD;
    }
  }

  // ---- stage 2: association  assoc = tanh([edge|te|self|other] @ W) ----
  gemm2(actS, actD, w_assoc, ASSOC_KMIN, ASSOC_KMIN + ASSOC_KWIN, ASSOC_KWIN, mrow, half, cS, cD);
#pragma unroll
  for (int nb = 0; nb < 8; ++nb) {
    int col = nb * 16 + mrow;
    float bv = assoc_b[col];
#pragma unroll
    for (int v = 0; v < 8; ++v) {
      int ro = v + half * 8;
      actS[ro * PITCH + OFF_A + col] = (h16)fast_tanh(cS[nb][v] + bv);
      actD[ro * PITCH + OFF_A + col] = (h16)fast_tanh(cD[nb][v] + bv);
    }
  }

  // ---- stage 3: comm layer 1  h = relu([edge|te|assoc] @ W1) ----
  gemm2(actS, actD, w_c1, C1_KMIN, C1_KMIN + C1_KWIN, C1_KWIN, mrow, half, cS, cD);
#pragma unroll
  for (int nb = 0; nb < 8; ++nb) {
    int col = nb * 16 + mrow;
    float bv = comm_b1[col];
#pragma unroll
    for (int v = 0; v < 8; ++v) {
      int ro = v + half * 8;
      float hS = cS[nb][v] + bv; hS = hS > 0.0f ? hS : 0.0f;
      float hD = cD[nb][v] + bv; hD = hD > 0.0f ? hD : 0.0f;
      actS[ro * PITCH + OFF_A + col] = (h16)hS;
      actD[ro * PITCH + OFF_A + col] = (h16)hD;
    }
  }

  // ---- stage 4: comm layer 2 + upd = evo + tanh(h @ W2 + b2) ----
  gemm2(actS, actD, w_c2, C2_KMIN, C2_KMIN + C2_KWIN, C2_KWIN, mrow, half, cS, cD);
#pragma unroll
  for (int nb = 0; nb < 8; ++nb) {
    int col = nb * 16 + mrow;
    float bv = comm_b2[col];
#pragma unroll
    for (int v = 0; v < 8; ++v) {
      int ro = v + half * 8;
      float evoS = (float)actD[ro * PITCH + OFF_B + col];  // evo_src persisted in dst.slotB
      float evoD = (float)actS[ro * PITCH + OFF_B + col];  // evo_dst persisted in src.slotB
      actS[ro * PITCH + OFF_A + col] = (h16)(fast_tanh(cS[nb][v] + bv) + evoS);
      actD[ro * PITCH + OFF_A + col] = (h16)(fast_tanh(cD[nb][v] + bv) + evoD);
    }
  }

  // node embeddings -> slot B (overwrites evo, which is no longer needed)
  for (int q = lane; q < ROWS_W * (MEMD / 4); q += 32) {
    int r  = q >> 5;
    int c4 = (q & 31) * 4;
    v4f es = *(const v4f*)(src_emb + (size_t)(row0 + r) * MEMD + c4);
    v4f ed = *(const v4f*)(dst_emb + (size_t)(row0 + r) * MEMD + c4);
    *(v4h*)(actS + r * PITCH + OFF_B + c4) = cvt4(es);
    *(v4h*)(actD + r * PITCH + OFF_B + c4) = cvt4(ed);
  }

  // ---- stage 5: output  out = [upd|emb] @ Wout + b ----
  gemm2(actS, actD, w_out, OUT_KMIN, OUT_KMIN + OUT_KWIN, OUT_KWIN, mrow, half, cS, cD);
#pragma unroll
  for (int nb = 0; nb < 8; ++nb) {
    int col = nb * 16 + mrow;
    float bv = out_b[col];
#pragma unroll
    for (int v = 0; v < 8; ++v) {
      int ro = v + half * 8;
      out[(size_t)(row0 + ro) * MEMD + col] = cS[nb][v] + bv;
      out[(size_t)(BATCH + row0 + ro) * MEMD + col] = cD[nb][v] + bv;
    }
  }
}

extern "C" void kernel_launch(void* const* d_in, const int* in_sizes, int n_in,
                              void* d_out, int out_size, void* d_ws, size_t ws_size,
                              hipStream_t stream) {
  const float* src_emb  = (const float*)d_in[0];
  const float* dst_emb  = (const float*)d_in[1];
  const int*   src_ids  = (const int*)  d_in[2];
  const int*   dst_ids  = (const int*)  d_in[3];
  const float* edge     = (const float*)d_in[4];
  const float* ts       = (const float*)d_in[5];
  const float* memory   = (const float*)d_in[6];
  const float* time_w   = (const float*)d_in[7];
  const float* time_b   = (const float*)d_in[8];
  const float* evo_w    = (const float*)d_in[9];
  const float* evo_b    = (const float*)d_in[10];
  const float* assoc_w  = (const float*)d_in[11];
  const float* assoc_b  = (const float*)d_in[12];
  const float* comm_w1  = (const float*)d_in[13];
  const float* comm_b1  = (const float*)d_in[14];
  const float* comm_w2  = (const float*)d_in[15];
  const float* comm_b2  = (const float*)d_in[16];
  const float* out_w    = (const float*)d_in[17];
  const float* out_b    = (const float*)d_in[18];
  float* out = (float*)d_out;

  h16* ws      = (h16*)d_ws;                 // needs 409,600 B of workspace
  h16* w_evo   = ws + WS_EVO;
  h16* w_assoc = ws + WS_ASSOC;
  h16* w_c1    = ws + WS_C1;
  h16* w_c2    = ws + WS_C2;
  h16* w_out   = ws + WS_OUT;

  Seg z{0, 0, 0};
  // evo_w rows: [mem 0..127 -> act 272..399], [te 128..227 -> act 172..271]
  pack_w<<<(128 * EVO_KWIN + 255) / 256, 256, 0, stream>>>(
      evo_w, w_evo, EVO_KWIN, EVO_KMIN, Seg{172, 272, 128}, Seg{272, 400, 0}, z, z);
  // assoc_w rows: [self 0..127], [other 128..255], [edge 256..427], [te 428..527]
  pack_w<<<(128 * ASSOC_KWIN + 255) / 256, 256, 0, stream>>>(
      assoc_w, w_assoc, ASSOC_KWIN, ASSOC_KMIN,
      Seg{0, 172, 256}, Seg{172, 272, 428}, Seg{272, 400, 0}, Seg{400, 528, 128});
  // comm_w1 rows: [assoc 0..127], [edge 128..299], [te 300..399]
  pack_w<<<(128 * C1_KWIN + 255) / 256, 256, 0, stream>>>(
      comm_w1, w_c1, C1_KWIN, C1_KMIN,
      Seg{0, 172, 128}, Seg{172, 272, 300}, Seg{272, 400, 0}, z);
  pack_w<<<(128 * C2_KWIN + 255) / 256, 256, 0, stream>>>(
      comm_w2, w_c2, C2_KWIN, C2_KMIN, Seg{272, 400, 0}, z, z, z);
  // out_w rows: [upd 0..127], [emb 128..255]
  pack_w<<<(128 * OUT_KWIN + 255) / 256, 256, 0, stream>>>(
      out_w, w_out, OUT_KWIN, OUT_KMIN, Seg{272, 400, 0}, Seg{400, 528, 128}, z, z);

  dim3 grid(BATCH / (ROWS_W * WPB));              // 2048 blocks
  size_t shmem = (size_t)WPB * 2 * ACT_H * sizeof(h16);  // 147,456 B
  dyrep_main<<<grid, WPB * 32, shmem, stream>>>(
      src_emb, dst_emb, src_ids, dst_ids, edge, ts, memory,
      time_w, time_b, evo_b, assoc_b, comm_b1, comm_b2, out_b,
      w_evo, w_assoc, w_c1, w_c2, w_out, out);
}
